// MultiHeadSelfAttention_74208444940646
// MI455X (gfx1250) — matrix operand order repaired and checked by reading, not compiled, not run on hardware
//
#include <hip/hip_runtime.h>
#include <hip/hip_bf16.h>
#include <hip/hip_fp16.h>

typedef __attribute__((ext_vector_type(16))) _Float16 v16h;
typedef __attribute__((ext_vector_type(8)))  _Float16 v8h;
typedef __attribute__((ext_vector_type(4)))  _Float16 v4h;
typedef __attribute__((ext_vector_type(8)))  float    v8f;
typedef __attribute__((ext_vector_type(4)))  unsigned int v4u;
typedef __attribute__((ext_vector_type(8)))  unsigned int v8u;

#define B_   4
#define S_   2048
#define D_   768
#define H_   12
#define HD_  64
#define MR_  (B_ * S_)      // 8192 rows
#define NEGV 1e10f
#define EPSV 1e-5f

// ---- LDS byte-offset of a __shared__ pointer (addrspace(3) ptrtoint) --------
typedef __attribute__((address_space(3))) const void lds_cv;
__device__ __forceinline__ unsigned lds_off(const void* p) {
  return (unsigned)(unsigned long long)(lds_cv*)p;
}

// ---- CDNA5 async copy: global -> LDS, 16B per lane, ASYNCcnt-tracked --------
__device__ __forceinline__ void async_b128(unsigned lds_byte, const _Float16* g) {
  asm volatile("global_load_async_to_lds_b128 %0, %1, off"
               :: "v"(lds_byte), "v"((unsigned long long)(uintptr_t)g)
               : "memory");
}

// ---- CDNA5 Tensor Data Mover: contiguous 1-D copy of nhalfs f16 -> LDS ------
// D# per ISA 8.3/8.4: group0 = {count=1,type=2,lds_addr,global_addr},
// group1 = {data_size=2B, tensor_dim0=nhalfs, tensor_dim1=1, tile_dim0=nhalfs}
__device__ __forceinline__ void tdm_copy_1d(unsigned lds_byte, const void* g,
                                            unsigned nhalfs) {
  unsigned long long ga = (unsigned long long)(uintptr_t)g;
  v4u g0;
  g0[0] = 1u;                                        // count=1, user mode
  g0[1] = lds_byte;                                  // lds_addr
  g0[2] = (unsigned)ga;                              // global_addr[31:0]
  g0[3] = ((unsigned)(ga >> 32) & 0x1FFFFFFu) | (2u << 30);  // addr[56:32], type=2
  v8u g1;
  g1[0] = 1u << 16;                                  // data_size=1 (2 bytes)
  g1[1] = (nhalfs & 0xFFFFu) << 16;                  // tensor_dim0[15:0]
  g1[2] = (nhalfs >> 16) | (1u << 16);               // tensor_dim0[31:16], tensor_dim1=1
  g1[3] = (nhalfs & 0xFFFFu) << 16;                  // tile_dim0
  g1[4] = 0u;                                        // tile_dim1=0 (1-D), tile_dim2=0
  g1[5] = nhalfs;                                    // tensor_dim0_stride[31:0]
  g1[6] = 0u;
  g1[7] = 0u;
  asm volatile("tensor_load_to_lds %0, %1" :: "s"(g0), "s"(g1) : "memory");
}

// ---- WMMA fragment helpers (contiguous k: element s of lane half h is k = 16h + s) -------------------
__device__ __forceinline__ v16h load_a_frag(const _Float16* __restrict__ tile, int stride) {
  const int lane = threadIdx.x & 31;
  const int row  = lane & 15, hf = lane >> 4;
  const _Float16* p = tile + row * stride + hf * 16;
  v8h lo = *(const v8h*)(p);
  v8h hi = *(const v8h*)(p + 8);
  v16h a;
#pragma unroll
  for (int i = 0; i < 8; ++i) { a[i] = lo[i]; a[8 + i] = hi[i]; }
  return a;
}
__device__ __forceinline__ v16h load_b_frag(const _Float16* __restrict__ tileT, int stride) {
  const int lane = threadIdx.x & 31;
  const int n = lane & 15, hf = lane >> 4;
  return *(const v16h*)(tileT + n * stride + hf * 16);
}
__device__ __forceinline__ v8f wmma_f16(v16h a, v16h b, v8f c) {
  return __builtin_amdgcn_wmma_f32_16x16x32_f16(false, a, false, b, (short)0, c,
                                                false, false);
}

// ---- Kernel 0a: f32 -> f16 straight conversion (context) --------------------
__global__ __launch_bounds__(256) void convert_kernel(
    const float* __restrict__ src, _Float16* __restrict__ dst) {
  size_t i = ((size_t)blockIdx.x * 256 + threadIdx.x) * 4;
  float4 f = *(const float4*)(src + i);
  v4h h; h[0] = (_Float16)f.x; h[1] = (_Float16)f.y;
  h[2] = (_Float16)f.z; h[3] = (_Float16)f.w;
  *(v4h*)(dst + i) = h;
}

// ---- Kernel 0b: f32 [R][C] -> f16 transposed [C][R] via LDS tile ------------
__global__ __launch_bounds__(256) void convert_transpose_kernel(
    const float* __restrict__ src, _Float16* __restrict__ dstT, int R, int C) {
  __shared__ float tile[64][65];
  const int r0 = blockIdx.y * 64, c0 = blockIdx.x * 64;
  const int tid = threadIdx.x;
#pragma unroll
  for (int i = 0; i < 16; ++i) {
    int idx = tid + i * 256, row = idx >> 6, col = idx & 63;
    tile[row][col] = src[(size_t)(r0 + row) * C + c0 + col];
  }
  __syncthreads();
#pragma unroll
  for (int i = 0; i < 16; ++i) {
    int idx = tid + i * 256, row = idx >> 6, col = idx & 63;
    dstT[(size_t)(c0 + row) * R + r0 + col] = (_Float16)tile[col][row];
  }
}

// ---- shared GEMM tile issue: A[128][32] + B[128][32] f16, rows stride 768 ---
__device__ __forceinline__ void issue_tile_async(
    const _Float16* __restrict__ gA, const _Float16* __restrict__ gB,
    unsigned aBase, unsigned bBase, int tid) {
#pragma unroll
  for (int i = 0; i < 2; ++i) {
    int idx = tid + i * 256, row = idx >> 2, c = idx & 3;
    async_b128(aBase + row * 64 + c * 16, gA + (size_t)row * D_ + c * 8);
  }
#pragma unroll
  for (int i = 0; i < 2; ++i) {
    int idx = tid + i * 256, row = idx >> 2, c = idx & 3;
    async_b128(bBase + row * 64 + c * 16, gB + (size_t)row * D_ + c * 8);
  }
}

// ---- Kernel 1: QKV GEMM (f16 A from ctx16, f16 B from wqkvT[N][K]) ----------
__global__ __launch_bounds__(256) void qkv_gemm_kernel(
    const _Float16* __restrict__ ctx16, const _Float16* __restrict__ wqkvT,
    _Float16* __restrict__ qbuf, _Float16* __restrict__ kbuf,
    _Float16* __restrict__ vbuf) {
  __shared__ __align__(32) _Float16 As[2][128][32];
  __shared__ __align__(32) _Float16 Bt[2][128][32];

  const int tid = threadIdx.x;
  const int lane = tid & 31, wave = tid >> 5;
  const int wm = wave >> 1, wn = wave & 1;
  const int m0 = blockIdx.x * 128, n0 = blockIdx.y * 128;

  const unsigned aB[2] = {lds_off(&As[0][0][0]), lds_off(&As[1][0][0])};
  const unsigned bB[2] = {lds_off(&Bt[0][0][0]), lds_off(&Bt[1][0][0])};
  const _Float16* gA = ctx16 + (size_t)m0 * D_;
  const _Float16* gB = wqkvT + (size_t)n0 * D_;

  v8f zero = {};
  v8f acc[2][4];
#pragma unroll
  for (int i = 0; i < 2; ++i)
#pragma unroll
    for (int j = 0; j < 4; ++j) acc[i][j] = zero;

  issue_tile_async(gA, gB, aB[0], bB[0], tid);
  for (int t = 0; t < D_ / 32; ++t) {
    const int cur = t & 1;
    if (t + 1 < D_ / 32) {
      issue_tile_async(gA + (t + 1) * 32, gB + (t + 1) * 32, aB[cur ^ 1],
                       bB[cur ^ 1], tid);
      asm volatile("s_wait_asynccnt 0x4" ::: "memory");
    } else {
      asm volatile("s_wait_asynccnt 0x0" ::: "memory");
    }
    __syncthreads();
    const _Float16* Ap = &As[cur][0][0];
    const _Float16* Bp = &Bt[cur][0][0];
    v16h af[2], bf[4];
#pragma unroll
    for (int tm = 0; tm < 2; ++tm) af[tm] = load_a_frag(Ap + (wm * 32 + tm * 16) * 32, 32);
#pragma unroll
    for (int tn = 0; tn < 4; ++tn) bf[tn] = load_b_frag(Bp + (wn * 64 + tn * 16) * 32, 32);
#pragma unroll
    for (int tm = 0; tm < 2; ++tm)
#pragma unroll
      for (int tn = 0; tn < 4; ++tn) acc[tm][tn] = wmma_f16(af[tm], bf[tn], acc[tm][tn]);
    __syncthreads();
  }

  const int hf = lane >> 4, nl = lane & 15;
#pragma unroll
  for (int tm = 0; tm < 2; ++tm) {
#pragma unroll
    for (int tn = 0; tn < 4; ++tn) {
      int col = n0 + wn * 64 + tn * 16 + nl;
      int which = col / D_;
      int c = col - which * D_;
      int h = c >> 6, hd = c & 63;
      _Float16* dst = (which == 0) ? qbuf : ((which == 1) ? kbuf : vbuf);
      float scale = (which == 0) ? 0.125f : 1.0f;   // fold 1/sqrt(64) into Q
#pragma unroll
      for (int v = 0; v < 8; ++v) {
        int row = m0 + wm * 32 + tm * 16 + hf * 8 + v;
        int b = row >> 11, s = row & 2047;
        dst[(((size_t)(b * H_ + h)) * S_ + s) * HD_ + hd] =
            (_Float16)(acc[tm][tn][v] * scale);
      }
    }
  }
}

// ---- Kernel 2: flash attention; Q/K staged by TDM, V transposed manually ----
__global__ __launch_bounds__(128) void attn_kernel(
    const _Float16* __restrict__ qbuf, const _Float16* __restrict__ kbuf,
    const _Float16* __restrict__ vbuf, const float* __restrict__ pad,
    _Float16* __restrict__ obuf) {
  __shared__ __align__(32) _Float16 Qs[64][64];
  __shared__ __align__(32) _Float16 Ks[64][64];   // row-major [k][d]
  __shared__ __align__(32) _Float16 Vt[64][64];   // transposed [d][k]
  __shared__ __align__(32) _Float16 Ps[4][16][64];

  const int tid = threadIdx.x, lane = tid & 31, wave = tid >> 5;
  const int bid = blockIdx.x;
  const int qt = bid & 31;
  const int h  = (bid >> 5) % H_;
  const int b  = bid / (32 * H_);
  const int q0 = qt * 64;
  const size_t hoff = ((size_t)(b * H_ + h)) * S_ * HD_;

  if (wave == 0) {                                // Q tile: one 8KB TDM copy
    tdm_copy_1d(lds_off(&Qs[0][0]), qbuf + hoff + (size_t)q0 * HD_, 64 * HD_);
    asm volatile("s_wait_tensorcnt 0x0" ::: "memory");
  }

  float mstate[8], lstate[8];
  v8f zero = {};
  v8f o[4];
#pragma unroll
  for (int v = 0; v < 8; ++v) { mstate[v] = -1e30f; lstate[v] = 0.f; }
#pragma unroll
  for (int n = 0; n < 4; ++n) o[n] = zero;

  const int hf = lane >> 4, nl = lane & 15;

  for (int kb = 0; kb < S_; kb += 64) {
    __syncthreads();
    if (wave == 0)                                // K chunk: one 8KB TDM copy
      tdm_copy_1d(lds_off(&Ks[0][0]), kbuf + hoff + (size_t)kb * HD_, 64 * HD_);
#pragma unroll
    for (int i = 0; i < 4; ++i) {                 // V chunk: stage transposed
      int idx = tid + i * 128;
      int row = idx >> 3, seg = (idx & 7) * 8;
      v8h vv = *(const v8h*)(vbuf + hoff + (size_t)(kb + row) * HD_ + seg);
#pragma unroll
      for (int j = 0; j < 8; ++j) Vt[seg + j][row] = vv[j];
    }
    if (wave == 0) asm volatile("s_wait_tensorcnt 0x0" ::: "memory");
    __syncthreads();

    v8f s[4];
#pragma unroll
    for (int n = 0; n < 4; ++n) s[n] = zero;
#pragma unroll
    for (int d2 = 0; d2 < 2; ++d2) {              // scores = Q K^T
      v16h a = load_a_frag(&Qs[wave * 16][0] + d2 * 32, 64);
#pragma unroll
      for (int n = 0; n < 4; ++n) {
        v16h bb = load_b_frag(&Ks[n * 16][0] + d2 * 32, 64);
        s[n] = wmma_f16(a, bb, s[n]);
      }
    }
#pragma unroll
    for (int n = 0; n < 4; ++n) {                 // pad mask
      float mv = pad[b * S_ + kb + n * 16 + nl];
#pragma unroll
      for (int v = 0; v < 8; ++v) s[n][v] = s[n][v] * mv - (1.f - mv) * NEGV;
    }
#pragma unroll
    for (int v = 0; v < 8; ++v) {                 // online softmax
      float x = fmaxf(fmaxf(s[0][v], s[1][v]), fmaxf(s[2][v], s[3][v]));
#pragma unroll
      for (int off = 8; off >= 1; off >>= 1) x = fmaxf(x, __shfl_xor(x, off, 32));
      float mnew = fmaxf(mstate[v], x);
      float corr = __expf(mstate[v] - mnew);
      mstate[v] = mnew;
#pragma unroll
      for (int n = 0; n < 4; ++n) s[n][v] = __expf(s[n][v] - mnew);
      float r = s[0][v] + s[1][v] + s[2][v] + s[3][v];
#pragma unroll
      for (int off = 8; off >= 1; off >>= 1) r += __shfl_xor(r, off, 32);
      lstate[v] = lstate[v] * corr + r;
#pragma unroll
      for (int n = 0; n < 4; ++n) o[n][v] *= corr;
    }
#pragma unroll
    for (int n = 0; n < 4; ++n)                   // C-layout -> A-layout via LDS
#pragma unroll
      for (int v = 0; v < 8; ++v)
        Ps[wave][hf * 8 + v][n * 16 + nl] = (_Float16)s[n][v];
    asm volatile("s_wait_dscnt 0x0" ::: "memory");
#pragma unroll
    for (int ks = 0; ks < 2; ++ks) {              // O += P V
      v16h a = load_a_frag(&Ps[wave][0][0] + ks * 32, 64);
#pragma unroll
      for (int n = 0; n < 4; ++n) {
        v16h bb = load_b_frag(&Vt[n * 16][0] + ks * 32, 64);
        o[n] = wmma_f16(a, bb, o[n]);
      }
    }
  }

#pragma unroll
  for (int n = 0; n < 4; ++n) {
#pragma unroll
    for (int v = 0; v < 8; ++v) {
      int q = q0 + wave * 16 + hf * 8 + v;
      float inv = 1.f / lstate[v];
      obuf[((size_t)(b * S_) + q) * D_ + h * HD_ + n * 16 + nl] =
          (_Float16)(o[n][v] * inv);
    }
  }
}

// ---- Kernel 3: out-proj GEMM + bias + residual -> X f32 ---------------------
__global__ __launch_bounds__(256) void proj_gemm_kernel(
    const _Float16* __restrict__ obuf, const _Float16* __restrict__ wprojT,
    const float* __restrict__ bproj, const float* __restrict__ ctx,
    float* __restrict__ X) {
  __shared__ __align__(32) _Float16 As[2][128][32];
  __shared__ __align__(32) _Float16 Bt[2][128][32];

  const int tid = threadIdx.x;
  const int lane = tid & 31, wave = tid >> 5;
  const int wm = wave >> 1, wn = wave & 1;
  const int m0 = blockIdx.x * 128, n0 = blockIdx.y * 128;

  const unsigned aB[2] = {lds_off(&As[0][0][0]), lds_off(&As[1][0][0])};
  const unsigned bB[2] = {lds_off(&Bt[0][0][0]), lds_off(&Bt[1][0][0])};
  const _Float16* gA = obuf + (size_t)m0 * D_;
  const _Float16* gB = wprojT + (size_t)n0 * D_;

  v8f zero = {};
  v8f acc[2][4];
#pragma unroll
  for (int i = 0; i < 2; ++i)
#pragma unroll
    for (int j = 0; j < 4; ++j) acc[i][j] = zero;

  issue_tile_async(gA, gB, aB[0], bB[0], tid);
  for (int t = 0; t < D_ / 32; ++t) {
    const int cur = t & 1;
    if (t + 1 < D_ / 32) {
      issue_tile_async(gA + (t + 1) * 32, gB + (t + 1) * 32, aB[cur ^ 1],
                       bB[cur ^ 1], tid);
      asm volatile("s_wait_asynccnt 0x4" ::: "memory");
    } else {
      asm volatile("s_wait_asynccnt 0x0" ::: "memory");
    }
    __syncthreads();
    const _Float16* Ap = &As[cur][0][0];
    const _Float16* Bp = &Bt[cur][0][0];
    v16h af[2], bf[4];
#pragma unroll
    for (int tm = 0; tm < 2; ++tm) af[tm] = load_a_frag(Ap + (wm * 32 + tm * 16) * 32, 32);
#pragma unroll
    for (int tn = 0; tn < 4; ++tn) bf[tn] = load_b_frag(Bp + (wn * 64 + tn * 16) * 32, 32);
#pragma unroll
    for (int tm = 0; tm < 2; ++tm)
#pragma unroll
      for (int tn = 0; tn < 4; ++tn) acc[tm][tn] = wmma_f16(af[tm], bf[tn], acc[tm][tn]);
    __syncthreads();
  }

  const int hf = lane >> 4, nl = lane & 15;
#pragma unroll
  for (int tm = 0; tm < 2; ++tm) {
#pragma unroll
    for (int tn = 0; tn < 4; ++tn) {
      int col = n0 + wn * 64 + tn * 16 + nl;
      float bias = bproj[col];
#pragma unroll
      for (int v = 0; v < 8; ++v) {
        int row = m0 + wm * 32 + tm * 16 + hf * 8 + v;
        size_t off = (size_t)row * D_ + col;
        X[off] = acc[tm][tn][v] + bias + ctx[off];   // + residual
      }
    }
  }
}

// ---- Kernel 4: LayerNorm, one 256-thread block per row ----------------------
__global__ __launch_bounds__(256) void ln_kernel(
    const float* __restrict__ X, const float* __restrict__ gamma,
    const float* __restrict__ beta, float* __restrict__ out) {
  __shared__ float red[16];
  const int row = blockIdx.x;
  const float* x = X + (size_t)row * D_;
  float v0 = x[threadIdx.x], v1 = x[threadIdx.x + 256], v2 = x[threadIdx.x + 512];
  float s = v0 + v1 + v2;
  float s2 = v0 * v0 + v1 * v1 + v2 * v2;
#pragma unroll
  for (int off = 16; off >= 1; off >>= 1) {
    s += __shfl_xor(s, off, 32);
    s2 += __shfl_xor(s2, off, 32);
  }
  int lane = threadIdx.x & 31, wave = threadIdx.x >> 5;
  if (lane == 0) { red[wave] = s; red[8 + wave] = s2; }
  __syncthreads();
  if (wave == 0) {
    float a = (lane < 8) ? red[lane] : 0.f;
    float c = (lane < 8) ? red[8 + lane] : 0.f;
#pragma unroll
    for (int off = 4; off >= 1; off >>= 1) {
      a += __shfl_xor(a, off, 32);
      c += __shfl_xor(c, off, 32);
    }
    if (lane == 0) { red[0] = a; red[1] = c; }
  }
  __syncthreads();
  float mu = red[0] * (1.f / D_);
  float var = red[1] * (1.f / D_) - mu * mu;
  float rs = rsqrtf(var + EPSV);
  float* o = out + (size_t)row * D_;
  o[threadIdx.x]       = (v0 - mu) * rs * gamma[threadIdx.x]       + beta[threadIdx.x];
  o[threadIdx.x + 256] = (v1 - mu) * rs * gamma[threadIdx.x + 256] + beta[threadIdx.x + 256];
  o[threadIdx.x + 512] = (v2 - mu) * rs * gamma[threadIdx.x + 512] + beta[threadIdx.x + 512];
}

extern "C" void kernel_launch(void* const* d_in, const int* in_sizes, int n_in,
                              void* d_out, int out_size, void* d_ws, size_t ws_size,
                              hipStream_t stream) {
  const float* ctx   = (const float*)d_in[0];
  const float* pad   = (const float*)d_in[1];
  const float* wqkv  = (const float*)d_in[2];
  const float* wproj = (const float*)d_in[3];
  const float* bproj = (const float*)d_in[4];
  const float* gamma = (const float*)d_in[5];
  const float* beta  = (const float*)d_in[6];

  const size_t elems = (size_t)MR_ * D_;          // 6,291,456
  _Float16* qbuf   = (_Float16*)d_ws;
  _Float16* kbuf   = qbuf + elems;
  _Float16* vbuf   = kbuf + elems;
  _Float16* obuf   = vbuf + elems;
  _Float16* ctx16  = obuf + elems;
  _Float16* wqkvT  = ctx16 + elems;               // [2304][768] f16
  _Float16* wprojT = wqkvT + (size_t)(3 * D_) * D_;  // [768][768] f16
  float*    X      = (float*)(wprojT + (size_t)D_ * D_);

  convert_kernel<<<dim3(elems / 1024), 256, 0, stream>>>(ctx, ctx16);
  convert_transpose_kernel<<<dim3((3 * D_) / 64, D_ / 64), 256, 0, stream>>>(
      wqkv, wqkvT, D_, 3 * D_);
  convert_transpose_kernel<<<dim3(D_ / 64, D_ / 64), 256, 0, stream>>>(
      wproj, wprojT, D_, D_);
  qkv_gemm_kernel<<<dim3(MR_ / 128, (3 * D_) / 128), 256, 0, stream>>>(
      ctx16, wqkvT, qbuf, kbuf, vbuf);
  attn_kernel<<<dim3(B_ * H_ * (S_ / 64)), 128, 0, stream>>>(
      qbuf, kbuf, vbuf, pad, obuf);
  proj_gemm_kernel<<<dim3(MR_ / 128, D_ / 128), 256, 0, stream>>>(
      obuf, wprojT, bproj, ctx, X);
  ln_kernel<<<dim3(MR_), 256, 0, stream>>>(X, gamma, beta, (float*)d_out);
}
